// NNConv_22127671509068
// MI455X (gfx1250) — matrix operand order repaired and checked by reading, not compiled, not run on hardware
//
#include <hip/hip_runtime.h>
#include <hip/hip_bf16.h>

typedef __attribute__((ext_vector_type(16))) _Float16 v16h;
typedef __attribute__((ext_vector_type(8)))  _Float16 v8h;
typedef __attribute__((ext_vector_type(2)))  _Float16 h2;
typedef __attribute__((ext_vector_type(8)))  float    v8f;

#define N_NODES   20000
#define N_EDGES   65536
#define WIDTH     64
#define EFEAT     6
#define HIDDEN    128
#define HID_PAD   130                  // 128 hidden + 1 bias row (h=1) + 1 zero pad
#define N_CHUNKS  (2*HID_PAD)          // flattened K = HID_PAD*64 = 8320 -> 260 chunks of 32
#define OUT_ELEMS (N_NODES*WIDTH)
#define EDGES_PER_WAVE 32              // two 16-row M-tiles per wave share B fragments

// ---------------- prep kernels ----------------

__global__ void zero_kernel(float* __restrict__ out, int n) {
  int i = blockIdx.x * blockDim.x + threadIdx.x;
  if (i < n) out[i] = 0.0f;
}

// h16[e, j] = relu(edge_attr[e] . W1[:, j] + b1[j])  (j<128), 1.0 (j==128), 0.0 (j==129)
__global__ void h_kernel(const float* __restrict__ ea, const float* __restrict__ W1,
                         const float* __restrict__ b1, _Float16* __restrict__ h16) {
  int idx = blockIdx.x * blockDim.x + threadIdx.x;
  if (idx >= N_EDGES * HID_PAD) return;
  int e = idx / HID_PAD, j = idx - e * HID_PAD;
  float v;
  if (j < HIDDEN) {
    float acc = b1[j];
#pragma unroll
    for (int i = 0; i < EFEAT; ++i) acc = fmaf(ea[e * EFEAT + i], W1[i * HIDDEN + j], acc);
    v = acc > 0.0f ? acc : 0.0f;
  } else {
    v = (j == HIDDEN) ? 1.0f : 0.0f;
  }
  h16[idx] = (_Float16)v;
}

// xs16[e, w] = (f16) x[senders[e], w]
__global__ void gather_kernel(const float* __restrict__ x, const int* __restrict__ senders,
                              _Float16* __restrict__ xs16) {
  int idx = blockIdx.x * blockDim.x + threadIdx.x;
  if (idx >= N_EDGES * WIDTH) return;
  int e = idx >> 6, w = idx & 63;
  xs16[idx] = (_Float16)x[(size_t)senders[e] * WIDTH + w];
}

// Reorder W2 (128 x 4096) into per-lane WMMA B fragments.
// Flattened K index: Kf = k*64 + w ; chunk c covers Kf = c*32 .. c*32+31 (k = c/2, w0 = (c&1)*32).
// B-fragment layout (f16 32x16): lane l<16 -> N=l, halves j=0..15 are K rows base+{0..7,16..23};
//                                lane l>=16 -> N=l-16, K rows base+{8..15,24..31}.
// bswz[c][nt][lane][j], nt = N-tile (v = nt*16 + N).
__global__ void bswz_kernel(const float* __restrict__ W2, const float* __restrict__ b2,
                            _Float16* __restrict__ bswz) {
  int idx = blockIdx.x * blockDim.x + threadIdx.x;
  if (idx >= N_CHUNKS * 4 * 32 * 16) return;
  int j  = idx & 15;
  int l  = (idx >> 4) & 31;
  int nt = (idx >> 9) & 3;
  int c  = idx >> 11;
  int t  = ((j & 8) << 1) + ((l >> 4) << 3) + (j & 7);
  int k  = c >> 1;
  int w  = ((c & 1) << 5) + t;
  int v  = nt * 16 + (l & 15);
  float val;
  if (k < HIDDEN)        val = W2[(size_t)k * (WIDTH * WIDTH) + w * WIDTH + v];
  else if (k == HIDDEN)  val = b2[w * WIDTH + v];
  else                   val = 0.0f;
  bswz[idx] = (_Float16)val;
}

// ---------------- main fused WMMA kernel ----------------
// 4 wave32 per block; each wave owns 32 edges = two 16-row M-tiles that SHARE
// every B fragment (16 WMMAs per 8 B loads -> ~32 flops per cache byte).
// msgs(32x64) = U(32x8320) @ W2'(8320x64); A built on the fly as h[m,k]*xs[m,w].
__global__ __launch_bounds__(128) void nnconv_wmma_kernel(
    const _Float16* __restrict__ h16,    // E x HID_PAD
    const _Float16* __restrict__ xs16,   // E x 64
    const _Float16* __restrict__ bswz,   // N_CHUNKS x 4 x 32 x 16 halves
    const int* __restrict__ receivers,
    float* __restrict__ out)             // N_NODES x 64, pre-zeroed
{
  __shared__ _Float16 lds_h[4][EDGES_PER_WAVE][HID_PAD];   // 33,280 B

  const int lane = threadIdx.x & 31;
  const int wave = threadIdx.x >> 5;
  const int tile = blockIdx.x * 4 + wave;
  const int e0   = tile * EDGES_PER_WAVE;
  const int row  = lane & 15;    // M row within each 16-row sub-tile
  const int hi   = lane >> 4;    // half-wave select

  // Stage the 32 x HID_PAD h-tile into LDS (contiguous: 32*65 dwords).
  {
    const unsigned int* src = (const unsigned int*)(h16 + (size_t)e0 * HID_PAD);
    unsigned int* dst = (unsigned int*)&lds_h[wave][0][0];
    for (int i = lane; i < EDGES_PER_WAVE * (HID_PAD / 2); i += 32) dst[i] = src[i];
  }

  // Per-lane xs slices for A fragments (ISA 16-bit A 16x32 striping):
  //   lane<16 needs w = {w0+0..7, w0+16..23}; lane>=16 needs {w0+8..15, w0+24..31}.
  // Two M-tiles -> two sets. xs*A: w0=0 chunks, xs*B: w0=32 chunks.
  v16h xs0A, xs0B, xs1A, xs1B;
  {
    const _Float16* xr0 = xs16 + (size_t)(e0 + row) * WIDTH + hi * 8;
    const _Float16* xr1 = xr0 + 16 * WIDTH;
    v8h a0 = *(const v8h*)(xr0 + 0);
    v8h a1 = *(const v8h*)(xr0 + 16);
    v8h a2 = *(const v8h*)(xr0 + 32);
    v8h a3 = *(const v8h*)(xr0 + 48);
    v8h b0 = *(const v8h*)(xr1 + 0);
    v8h b1 = *(const v8h*)(xr1 + 16);
    v8h b2 = *(const v8h*)(xr1 + 32);
    v8h b3 = *(const v8h*)(xr1 + 48);
#pragma unroll
    for (int i = 0; i < 8; ++i) {
      xs0A[i] = a0[i]; xs0A[8 + i] = a1[i];
      xs0B[i] = a2[i]; xs0B[8 + i] = a3[i];
      xs1A[i] = b0[i]; xs1A[8 + i] = b1[i];
      xs1B[i] = b2[i]; xs1B[8 + i] = b3[i];
    }
  }

  __syncthreads();

  v8f acc[8] = {};   // [0..3]: N-tiles for edges e0..e0+15, [4..7]: edges e0+16..e0+31

  const v16h* bp = (const v16h*)bswz;
  for (int k = 0; k < HID_PAD; ++k) {
    _Float16 hk0 = lds_h[wave][row][k];
    _Float16 hk1 = lds_h[wave][16 + row][k];
    h2 hv0; hv0[0] = hk0; hv0[1] = hk0;
    h2 hv1; hv1[0] = hk1; hv1[1] = hk1;

    // Build 4 A fragments (2 M-tiles x 2 chunks) via packed f16 muls.
    v16h a0A, a0B, a1A, a1B;
    {
      const h2* x0a = (const h2*)&xs0A;  const h2* x0b = (const h2*)&xs0B;
      const h2* x1a = (const h2*)&xs1A;  const h2* x1b = (const h2*)&xs1B;
      h2* p0a = (h2*)&a0A;  h2* p0b = (h2*)&a0B;
      h2* p1a = (h2*)&a1A;  h2* p1b = (h2*)&a1B;
#pragma unroll
      for (int i = 0; i < 8; ++i) {
        p0a[i] = hv0 * x0a[i];  p0b[i] = hv0 * x0b[i];
        p1a[i] = hv1 * x1a[i];  p1b[i] = hv1 * x1b[i];
      }
    }

    const v16h* bc0 = bp + (size_t)(2 * k) * 4 * 32 + lane;   // chunk 2k   (w0=0)
    const v16h* bc1 = bc0 + 4 * 32;                           // chunk 2k+1 (w0=32)
    __builtin_prefetch(bc0 + 8 * 32, 0, 0);                   // next k's B stream

#pragma unroll
    for (int nt = 0; nt < 4; ++nt) {
      v16h b = bc0[nt * 32];
      acc[nt]     = __builtin_amdgcn_wmma_f32_16x16x32_f16(
          false, a0A, false, b, (short)0, acc[nt],     false, false);
      acc[4 + nt] = __builtin_amdgcn_wmma_f32_16x16x32_f16(
          false, a1A, false, b, (short)0, acc[4 + nt], false, false);
    }
#pragma unroll
    for (int nt = 0; nt < 4; ++nt) {
      v16h b = bc1[nt * 32];
      acc[nt]     = __builtin_amdgcn_wmma_f32_16x16x32_f16(
          false, a0B, false, b, (short)0, acc[nt],     false, false);
      acc[4 + nt] = __builtin_amdgcn_wmma_f32_16x16x32_f16(
          false, a1B, false, b, (short)0, acc[4 + nt], false, false);
    }
  }

  // Scatter-add: C/D layout -> lane holds (M = j + 8*hi, N = lane&15) per VGPR j.
#pragma unroll
  for (int j = 0; j < 8; ++j) {
    int m    = j + 8 * hi;
    int rcv0 = receivers[e0 + m];
    int rcv1 = receivers[e0 + 16 + m];
    float* orow0 = out + (size_t)rcv0 * WIDTH + (lane & 15);
    float* orow1 = out + (size_t)rcv1 * WIDTH + (lane & 15);
#pragma unroll
    for (int nt = 0; nt < 4; ++nt) {
      atomicAdd(orow0 + nt * 16, acc[nt][j]);
      atomicAdd(orow1 + nt * 16, acc[4 + nt][j]);
    }
  }
}

// ---------------- launch ----------------
extern "C" void kernel_launch(void* const* d_in, const int* in_sizes, int n_in,
                              void* d_out, int out_size, void* d_ws, size_t ws_size,
                              hipStream_t stream) {
  const float* x         = (const float*)d_in[0];
  const float* edge_attr = (const float*)d_in[1];
  const float* W1        = (const float*)d_in[2];
  const float* b1        = (const float*)d_in[3];
  const float* W2        = (const float*)d_in[4];
  const float* b2        = (const float*)d_in[5];
  const int*   senders   = (const int*)d_in[6];
  const int*   receivers = (const int*)d_in[7];
  float* out = (float*)d_out;

  // workspace carve-out (all 256B aligned)
  char* ws = (char*)d_ws;
  size_t o = 0;
  _Float16* h16  = (_Float16*)(ws + o);
  o += (size_t)N_EDGES * HID_PAD * sizeof(_Float16); o = (o + 255) & ~(size_t)255;
  _Float16* xs16 = (_Float16*)(ws + o);
  o += (size_t)N_EDGES * WIDTH * sizeof(_Float16);   o = (o + 255) & ~(size_t)255;
  _Float16* bswz = (_Float16*)(ws + o);

  zero_kernel<<<(OUT_ELEMS + 255) / 256, 256, 0, stream>>>(out, OUT_ELEMS);
  h_kernel<<<(N_EDGES * HID_PAD + 255) / 256, 256, 0, stream>>>(edge_attr, W1, b1, h16);
  gather_kernel<<<(N_EDGES * WIDTH + 255) / 256, 256, 0, stream>>>(x, senders, xs16);
  bswz_kernel<<<(N_CHUNKS * 4 * 32 * 16 + 255) / 256, 256, 0, stream>>>(W2, b2, bswz);
  nnconv_wmma_kernel<<<N_EDGES / EDGES_PER_WAVE / 4, 128, 0, stream>>>(
      h16, xs16, bswz, receivers, out);
}
